// IthemalRNN_15582141349911
// MI455X (gfx1250) — 16-bit and 8-bit matrix operand order repaired and checked by reading, compiled, not run on hardware
//
#include <hip/hip_runtime.h>

#define NSEQ 4096
#define TLEN 16
#define HDIM 256
#define G4   1024
#define VOCAB 50000

typedef __attribute__((ext_vector_type(16))) __bf16 v16bf;
typedef __attribute__((ext_vector_type(8)))  float  v8f;
typedef __attribute__((ext_vector_type(8)))  int    v8i;

static __device__ __forceinline__ unsigned short f32_bf16(float f) {
  unsigned u = __builtin_bit_cast(unsigned, f);
  u += 0x7FFFu + ((u >> 16) & 1u);
  return (unsigned short)(u >> 16);
}

static __device__ __forceinline__ unsigned char f32_fp8(float x) {
  unsigned char s = 0;
  if (x < 0.f) { s = 0x80; x = -x; }
  if (!(x > 0.f)) return s;
  if (x >= 448.f) return (unsigned char)(s | 0x7E);
  int e; float m = frexpf(x, &e);
  int E = e + 6;
  if (E >= 1) {
    int mant = (int)(m * 16.f + 0.5f);
    if (mant >= 16) { mant = 8; ++E; }
    mant -= 8;
    if (E > 15 || (E == 15 && mant > 6)) return (unsigned char)(s | 0x7E);
    return (unsigned char)(s | (E << 3) | mant);
  }
  int mant = (int)(x * 512.f + 0.5f);
  if (mant > 7) mant = 7;
  return (unsigned char)(s | mant);
}

static __device__ __forceinline__ float sigf(float x) {
  return __builtin_amdgcn_rcpf(1.f + __expf(-x));
}
static __device__ __forceinline__ float tanh_fast(float x) {
  float cx = fminf(fmaxf(x, -10.f), 10.f);
  float e  = __expf(2.f * cx);
  return (e - 1.f) * __builtin_amdgcn_rcpf(e + 1.f);
}

union ABF { v16bf v; uint4 q[2]; };
union AF8 { v8i v; uint2 d[4]; };
union BF8 { v8i v; uint4 q[2]; };

static __device__ __forceinline__ void load_a_bf(ABF& a, const unsigned short* row, int lhalf) {
  a.q[0] = *reinterpret_cast<const uint4*>(row + lhalf * 8);
  a.q[1] = *reinterpret_cast<const uint4*>(row + 16 + lhalf * 8);
}
static __device__ __forceinline__ void load_b_bf(ABF& b, const unsigned short* wrow, int lhalf) {
  b.q[0] = *reinterpret_cast<const uint4*>(wrow + lhalf * 8);
  b.q[1] = *reinterpret_cast<const uint4*>(wrow + 16 + lhalf * 8);
}

#define R_EMB   (VOCAB * HDIM)
#define R_WCAT  (1024 * 512)
#define R_WINS  (1024 * 256)
#define R_WHH8  (1024 * 256)
#define R_ZF32  (3 * NSEQ * HDIM)
#define R_ZU16  (2 * NSEQ * HDIM)
#define R_TOTAL (R_EMB + R_WCAT + R_WINS + R_WHH8 + R_ZF32 + R_ZU16)

__global__ void prep_kernel(const float* __restrict__ emb,
                            const float* __restrict__ wih_tok, const float* __restrict__ whh_tok,
                            const float* __restrict__ wih_ins, const float* __restrict__ whh_ins,
                            unsigned short* __restrict__ embf, unsigned short* __restrict__ wcat,
                            unsigned short* __restrict__ wins, unsigned char* __restrict__ whh8,
                            float* __restrict__ zf32, unsigned short* __restrict__ zu16) {
  for (long i = blockIdx.x * blockDim.x + threadIdx.x; i < R_TOTAL; i += (long)gridDim.x * blockDim.x) {
    long j = i;
    if (j < R_EMB) { embf[j] = f32_bf16(emb[j]); continue; }
    j -= R_EMB;
    if (j < R_WCAT) {
      int n = (int)(j >> 9), k = (int)(j & 511);
      float f = (k < 256) ? wih_tok[n * 256 + k] : whh_tok[n * 256 + (k - 256)];
      wcat[j] = f32_bf16(f); continue;
    }
    j -= R_WCAT;
    if (j < R_WINS) { wins[j] = f32_bf16(wih_ins[j]); continue; }
    j -= R_WINS;
    if (j < R_WHH8) { whh8[j] = f32_fp8(whh_ins[j]); continue; }
    j -= R_WHH8;
    if (j < R_ZF32) { zf32[j] = 0.f; continue; }
    j -= R_ZF32;
    zu16[j] = 0;
  }
}

struct TokFrags { ABF a0, a1, b[4]; };

static __device__ __forceinline__ void tok_load(TokFrags& f, int k0,
    const unsigned short* e0, const unsigned short* e1,
    const unsigned short* h0p, const unsigned short* h1p,
    const unsigned short* wbase, int lhalf) {
  const unsigned short* s0 = (k0 < 256) ? (e0 + k0) : (h0p + (k0 - 256));
  const unsigned short* s1 = (k0 < 256) ? (e1 + k0) : (h1p + (k0 - 256));
  load_a_bf(f.a0, s0, lhalf);
  load_a_bf(f.a1, s1, lhalf);
#pragma unroll
  for (int g = 0; g < 4; ++g)
    load_b_bf(f.b[g], wbase + (size_t)g * 256 * 512 + k0, lhalf);
}

static __device__ __forceinline__ void lstm_epilogue(
    const v8f* acc, int mbase, int hcol, int lhalf, int t,
    float bi0, float bi1, float bi2, float bi3,
    const int* __restrict__ lengths, const float* __restrict__ h_in,
    float* __restrict__ h_out, unsigned short* __restrict__ hbf_out,
    float* __restrict__ c_buf) {
#pragma unroll
  for (int r = 0; r < 8; ++r) {
    const int mrow = mbase + lhalf * 8 + r;
    const float gi = sigf     (acc[0][r] + bi0);
    const float gf = sigf     (acc[1][r] + bi1);
    const float gg = tanh_fast(acc[2][r] + bi2);
    const float go = sigf     (acc[3][r] + bi3);
    const size_t idx = (size_t)mrow * HDIM + hcol;
    const float c_old = c_buf[idx];
    const float h_old = h_in[idx];
    const float c_new = gf * c_old + gi * gg;
    const float h_new = go * tanh_fast(c_new);
    const bool mk = t < lengths[mrow];
    const float hw = mk ? h_new : h_old;
    h_out[idx]   = hw;
    hbf_out[idx] = f32_bf16(hw);
    c_buf[idx]   = mk ? c_new : c_old;
  }
}

__global__ void __launch_bounds__(128)
tok_step_kernel(int t, const int* __restrict__ tokens, const int* __restrict__ lengths,
                const unsigned short* __restrict__ embf, const unsigned short* __restrict__ wcat,
                const float* __restrict__ b_ih, const float* __restrict__ b_hh,
                const float* __restrict__ h_in, const unsigned short* __restrict__ hbf_in,
                float* __restrict__ h_out, unsigned short* __restrict__ hbf_out,
                float* __restrict__ c_buf) {
  const int wid   = blockIdx.x * 4 + (threadIdx.x >> 5);
  const int lane  = threadIdx.x & 31;
  const int lrow  = lane & 15;
  const int lhalf = lane >> 4;
  const int tile_m = wid >> 4;
  const int tile_h = wid & 15;

  const int m0 = tile_m * 32 + lrow;
  const int m1 = m0 + 16;
  const int tok0 = tokens[(size_t)m0 * TLEN + t];
  const int tok1 = tokens[(size_t)m1 * TLEN + t];
  const unsigned short* e0  = embf + (size_t)tok0 * HDIM;
  const unsigned short* e1  = embf + (size_t)tok1 * HDIM;
  const unsigned short* h0p = hbf_in + (size_t)m0 * HDIM;
  const unsigned short* h1p = hbf_in + (size_t)m1 * HDIM;
  const unsigned short* wbase = wcat + (size_t)(tile_h * 16 + lrow) * 512;

  v8f zf = {0.f,0.f,0.f,0.f,0.f,0.f,0.f,0.f};
  v8f acc0[4] = {zf, zf, zf, zf};
  v8f acc1[4] = {zf, zf, zf, zf};

  TokFrags f[2];
  tok_load(f[0], 0, e0, e1, h0p, h1p, wbase, lhalf);
#pragma unroll
  for (int i = 0; i < 16; ++i) {
    if (i + 1 < 16)
      tok_load(f[(i + 1) & 1], (i + 1) * 32, e0, e1, h0p, h1p, wbase, lhalf);
    const TokFrags& c = f[i & 1];
#pragma unroll
    for (int g = 0; g < 4; ++g) {
      acc0[g] = __builtin_amdgcn_wmma_f32_16x16x32_bf16(false, c.a0.v, false, c.b[g].v,
                                                        (short)0, acc0[g], false, false);
      acc1[g] = __builtin_amdgcn_wmma_f32_16x16x32_bf16(false, c.a1.v, false, c.b[g].v,
                                                        (short)0, acc1[g], false, false);
    }
  }

  const int hcol = tile_h * 16 + lrow;
  const float bi0 = b_ih[hcol]       + b_hh[hcol];
  const float bi1 = b_ih[256 + hcol] + b_hh[256 + hcol];
  const float bi2 = b_ih[512 + hcol] + b_hh[512 + hcol];
  const float bi3 = b_ih[768 + hcol] + b_hh[768 + hcol];
  lstm_epilogue(acc0, tile_m * 32,      hcol, lhalf, t, bi0, bi1, bi2, bi3,
                lengths, h_in, h_out, hbf_out, c_buf);
  lstm_epilogue(acc1, tile_m * 32 + 16, hcol, lhalf, t, bi0, bi1, bi2, bi3,
                lengths, h_in, h_out, hbf_out, c_buf);
}

__global__ void __launch_bounds__(128)
ins_pre_kernel(const unsigned short* __restrict__ hbf, const unsigned short* __restrict__ wins,
               const float* __restrict__ b_ih_ins, float* __restrict__ xpi) {
  const int wid   = blockIdx.x * 4 + (threadIdx.x >> 5);
  const int lane  = threadIdx.x & 31;
  const int lrow  = lane & 15;
  const int lhalf = lane >> 4;
  const int tile_m = wid >> 6;
  const int tile_n = wid & 63;

  const unsigned short* hrow = hbf + (size_t)(tile_m * 16 + lrow) * HDIM;
  const int n = tile_n * 16 + lrow;
  const unsigned short* wrow = wins + (size_t)n * HDIM;

  v8f acc = {0.f,0.f,0.f,0.f,0.f,0.f,0.f,0.f};
#pragma unroll
  for (int i = 0; i < 8; ++i) {
    const int k0 = i * 32;
    ABF a, b;
    load_a_bf(a, hrow + k0, lhalf);
    load_b_bf(b, wrow + k0, lhalf);
    acc = __builtin_amdgcn_wmma_f32_16x16x32_bf16(false, a.v, false, b.v,
                                                  (short)0, acc, false, false);
  }
  const float bn = b_ih_ins[n];
#pragma unroll
  for (int r = 0; r < 8; ++r) {
    const int mrow = tile_m * 16 + lhalf * 8 + r;
    xpi[(size_t)mrow * G4 + n] = acc[r] + bn;
  }
}

__global__ void __launch_bounds__(512)
ins_scan_kernel(const float* __restrict__ xpi, const unsigned char* __restrict__ whh8,
                const float* __restrict__ b_hh_ins, const float* __restrict__ lin_w,
                const float* __restrict__ lin_b, float* __restrict__ out) {
  __shared__ float pre_lds[G4];
  __shared__ float hbuf[HDIM];
  __shared__ float bhh_lds[G4];
  __shared__ uint2 h8q[32];

  const int tid   = threadIdx.x;
  const int wave  = tid >> 5;
  const int lane  = tid & 31;
  const int lrow  = lane & 15;
  const int lhalf = lane >> 4;
  const int colbase = wave * 64;

  for (int i = tid; i < G4; i += 512) bhh_lds[i] = b_hh_ins[i];
  if (tid < HDIM) hbuf[tid] = 0.f;
  float creg = 0.f;

  BF8 bfr[4][4];
#pragma unroll
  for (int ct = 0; ct < 4; ++ct) {
    const int col = colbase + ct * 16 + lrow;
#pragma unroll
    for (int kc = 0; kc < 4; ++kc) {
      const uint4* bp = reinterpret_cast<const uint4*>(whh8 + (size_t)col * HDIM + kc * 64 + lhalf * 16);
      bfr[ct][kc].q[0] = bp[0];
      bfr[ct][kc].q[1] = bp[2];
    }
  }
  __syncthreads();

  for (int s = 0; s < NSEQ; ++s) {
    if (tid == 0 && s + 1 < NSEQ) __builtin_prefetch(xpi + (size_t)(s + 1) * G4, 0, 0);
    if (tid < HDIM) ((unsigned char*)h8q)[tid] = f32_fp8(hbuf[tid]);
    __syncthreads();

    AF8 afr[4];
    const uint2 z2 = {0u, 0u};
#pragma unroll
    for (int kc = 0; kc < 4; ++kc) {
      if (lrow == 0) {
        afr[kc].d[0] = h8q[kc * 8 + 0 + 2 * lhalf];
        afr[kc].d[1] = h8q[kc * 8 + 1 + 2 * lhalf];
        afr[kc].d[2] = h8q[kc * 8 + 4 + 2 * lhalf];
        afr[kc].d[3] = h8q[kc * 8 + 5 + 2 * lhalf];
      } else {
        afr[kc].d[0] = z2; afr[kc].d[1] = z2; afr[kc].d[2] = z2; afr[kc].d[3] = z2;
      }
    }
    v8f acc[4];
#pragma unroll
    for (int ct = 0; ct < 4; ++ct) { v8f zf = {0.f,0.f,0.f,0.f,0.f,0.f,0.f,0.f}; acc[ct] = zf; }
#pragma unroll
    for (int kc = 0; kc < 4; ++kc)
#pragma unroll
      for (int ct = 0; ct < 4; ++ct)
        acc[ct] = __builtin_amdgcn_wmma_f32_16x16x64_fp8_fp8(afr[kc].v, bfr[ct][kc].v,
                                                             (short)0, acc[ct], false, false);
    if (lane < 16) {
#pragma unroll
      for (int ct = 0; ct < 4; ++ct) pre_lds[colbase + ct * 16 + lane] = acc[ct][0];
    }
    __syncthreads();

    if (tid < HDIM) {
      const float* xr = xpi + (size_t)s * G4;
      const float gi = sigf     (pre_lds[tid]       + xr[tid]       + bhh_lds[tid]);
      const float gf = sigf     (pre_lds[256 + tid] + xr[256 + tid] + bhh_lds[256 + tid]);
      const float gg = tanh_fast(pre_lds[512 + tid] + xr[512 + tid] + bhh_lds[512 + tid]);
      const float go = sigf     (pre_lds[768 + tid] + xr[768 + tid] + bhh_lds[768 + tid]);
      creg = gf * creg + gi * gg;
      hbuf[tid] = go * tanh_fast(creg);
    }
    __syncthreads();
  }

  if (tid < HDIM) pre_lds[tid] = hbuf[tid] * lin_w[tid];
  __syncthreads();
  if (tid == 0) {
    float sm = 0.f;
    for (int k = 0; k < HDIM; ++k) sm += pre_lds[k];
    out[0] = sm + lin_b[0];
  }
}

extern "C" void kernel_launch(void* const* d_in, const int* in_sizes, int n_in,
                              void* d_out, int out_size, void* d_ws, size_t ws_size,
                              hipStream_t stream) {
  (void)in_sizes; (void)n_in; (void)out_size; (void)ws_size;
  const int*   tokens  = (const int*)  d_in[0];
  const int*   lengths = (const int*)  d_in[1];
  const float* emb     = (const float*)d_in[2];
  const float* wih_tok = (const float*)d_in[3];
  const float* whh_tok = (const float*)d_in[4];
  const float* bih_tok = (const float*)d_in[5];
  const float* bhh_tok = (const float*)d_in[6];
  const float* wih_ins = (const float*)d_in[7];
  const float* whh_ins = (const float*)d_in[8];
  const float* bih_ins = (const float*)d_in[9];
  const float* bhh_ins = (const float*)d_in[10];
  const float* lin_w   = (const float*)d_in[11];
  const float* lin_b   = (const float*)d_in[12];

  char* w = (char*)d_ws;
  unsigned short* embf = (unsigned short*)w; w += (size_t)VOCAB * HDIM * 2;
  unsigned short* wcat = (unsigned short*)w; w += (size_t)1024 * 512 * 2;
  unsigned short* wins = (unsigned short*)w; w += (size_t)1024 * 256 * 2;
  unsigned char*  whh8 = (unsigned char*) w; w += (size_t)1024 * 256;
  float* h0   = (float*)w;          w += (size_t)NSEQ * HDIM * 4;
  float* h1   = (float*)w;          w += (size_t)NSEQ * HDIM * 4;
  float* cb   = (float*)w;          w += (size_t)NSEQ * HDIM * 4;
  unsigned short* hbf0 = (unsigned short*)w; w += (size_t)NSEQ * HDIM * 2;
  unsigned short* hbf1 = (unsigned short*)w; w += (size_t)NSEQ * HDIM * 2;
  float* xpi  = (float*)w;

  prep_kernel<<<4096, 256, 0, stream>>>(emb, wih_tok, whh_tok, wih_ins, whh_ins,
                                        embf, wcat, wins, whh8, h0, hbf0);

  const float* hin = h0;           float* hout = h1;
  const unsigned short* hbin = hbf0; unsigned short* hbout = hbf1;
  for (int t = 0; t < TLEN; ++t) {
    tok_step_kernel<<<512, 128, 0, stream>>>(t, tokens, lengths, embf, wcat,
                                             bih_tok, bhh_tok, hin, hbin,
                                             hout, hbout, cb);
    const float* tf = hout; hout = (float*)hin; hin = tf;
    const unsigned short* tb = hbout; hbout = (unsigned short*)hbin; hbin = tb;
  }

  ins_pre_kernel<<<4096, 128, 0, stream>>>(hbin, wins, bih_ins, xpi);
  ins_scan_kernel<<<1, 512, 0, stream>>>(xpi, whh8, bhh_ins, lin_w, lin_b, (float*)d_out);
}
